// LeapRnnModel_63110249447902
// MI455X (gfx1250) — matrix operand order repaired and checked by reading, not compiled, not run on hardware
//
#include <hip/hip_runtime.h>

// =====================================================================
// LeapRnnModel forward for MI455X (gfx1250), wave32 + WMMA.
//  - All GEMMs use v_wmma_f32_16x16x32_f16 (f16 in, f32 accumulate).
//  - Per-step fused gates GEMM: gates = [x_t | h] @ [Wih | Whh]^T,
//    both LSTM directions in one launch (blockIdx.z), then a pointwise
//    LSTM cell kernel. Weights converted to f16 once per call (stay in L2).
// =====================================================================

typedef __attribute__((ext_vector_type(16))) _Float16 v16h;
typedef __attribute__((ext_vector_type(8)))  _Float16 v8h;
typedef __attribute__((ext_vector_type(8)))  float    v8f;

#define BB 1024
#define TT 60

// ---------------- A-fragment loader (16-bit A 16x32 ISA layout) -------
// lane<16 : row M=lane,    K = k0+0..7  and k0+16..23
// lane>=16: row M=lane-16, K = k0+8..15 and k0+24..31
__device__ __forceinline__ v16h load_a_frag(const _Float16* p) {
  v8h lo = *(const v8h*)(p);
  v8h hi = *(const v8h*)(p + 16);
  v16h r;
#pragma unroll
  for (int i = 0; i < 8; ++i) { r[i] = lo[i]; r[i + 8] = hi[i]; }
  return r;
}

// ---------------- WMMA GEMM ------------------------------------------
// C[m,n] = sum_k A0[m,k]*W0[n,k] + sum_k A1[m,k]*W1[n,k]
// A row-major f16 (lda in elements), W row-major f16 [N x K], C f32.
// Per-z (blockIdx.z) pointer offsets allow batching both LSTM directions.
// Block: 128 thr = 4 waves, tile 64(M) x 128(N); wave tile 32x64 = 8 WMMAs.
__global__ void __launch_bounds__(128)
wmma_gemm_kernel(const _Float16* __restrict__ A0, long long a0zOff, int lda0, int K0,
                 const _Float16* __restrict__ A1, long long a1zOff, int lda1, int K1,
                 const _Float16* __restrict__ W0, long long w0zOff,
                 const _Float16* __restrict__ W1, long long w1zOff,
                 float* __restrict__ C, long long czOff, int ldc)
{
  const int z     = blockIdx.z;
  const int lane  = threadIdx.x & 31;
  const int wave  = threadIdx.x >> 5;
  const int m0    = blockIdx.y * 64 + (wave >> 1) * 32;
  const int n0    = blockIdx.x * 128 + (wave & 1) * 64;
  const int lrow  = lane & 15;
  const int lhalf = lane >> 4;

  A0 += (long long)z * a0zOff;
  W0 += (long long)z * w0zOff;
  C  += (long long)z * czOff;

  v8f acc[2][4] = {};

  { // ---- operand 0: A0 x W0^T over K0 ----
    const _Float16* ap  = A0 + (long long)(m0 + lrow) * lda0 + lhalf * 8;
    const _Float16* bp  = W0 + (long long)(n0 + lrow) * (long long)K0 + lhalf * 8;
    const long long a16 = (long long)16 * lda0;
    for (int k = 0; k < K0; k += 32) {
      v16h aA = load_a_frag(ap + k);
      v16h aB = load_a_frag(ap + a16 + k);
#pragma unroll
      for (int j = 0; j < 4; ++j) {
        // B = W^T: lane n holds W[n0+16j+(n&15), k0 + (n>>4)*8 + {0..7, 16..23}]
        v16h bf = load_a_frag(bp + (long long)(16 * j) * K0 + k);
        acc[0][j] = __builtin_amdgcn_wmma_f32_16x16x32_f16(false, aA, false, bf,
                                                           (short)0, acc[0][j], false, false);
        acc[1][j] = __builtin_amdgcn_wmma_f32_16x16x32_f16(false, aB, false, bf,
                                                           (short)0, acc[1][j], false, false);
      }
      __builtin_prefetch((const void*)(bp + k + 32), 0, 0);  // global_prefetch next K-chunk of W
    }
  }

  if (K1 > 0) { // ---- operand 1: A1 x W1^T over K1 (recurrent h part) ----
    const _Float16* A1z = A1 + (long long)z * a1zOff;
    const _Float16* W1z = W1 + (long long)z * w1zOff;
    const _Float16* ap  = A1z + (long long)(m0 + lrow) * lda1 + lhalf * 8;
    const _Float16* bp  = W1z + (long long)(n0 + lrow) * (long long)K1 + lhalf * 8;
    const long long a16 = (long long)16 * lda1;
    for (int k = 0; k < K1; k += 32) {
      v16h aA = load_a_frag(ap + k);
      v16h aB = load_a_frag(ap + a16 + k);
#pragma unroll
      for (int j = 0; j < 4; ++j) {
        v16h bf = load_a_frag(bp + (long long)(16 * j) * K1 + k);
        acc[0][j] = __builtin_amdgcn_wmma_f32_16x16x32_f16(false, aA, false, bf,
                                                           (short)0, acc[0][j], false, false);
        acc[1][j] = __builtin_amdgcn_wmma_f32_16x16x32_f16(false, aB, false, bf,
                                                           (short)0, acc[1][j], false, false);
      }
    }
  }

  // ---- epilogue: C/D layout VGPR r -> M = r + 8*(lane>=16), N = lane&15 ----
  const int mrow = lhalf * 8;
#pragma unroll
  for (int i = 0; i < 2; ++i) {
#pragma unroll
    for (int j = 0; j < 4; ++j) {
      float* cp = C + (long long)(m0 + 16 * i + mrow) * ldc + (n0 + 16 * j + lrow);
#pragma unroll
      for (int r = 0; r < 8; ++r) cp[(long long)r * ldc] = acc[i][j][r];
    }
  }
}

// ---------------- LSTM pointwise cell --------------------------------
__device__ __forceinline__ float sigm(float x) { return 1.0f / (1.0f + __expf(-x)); }

__global__ void __launch_bounds__(256)
lstm_cell_kernel(const float* __restrict__ gates,                  // [2][B][2048]
                 const float* __restrict__ bih_f, const float* __restrict__ bhh_f,
                 const float* __restrict__ bih_b, const float* __restrict__ bhh_b,
                 float* __restrict__ c_state,                      // [2][B][512] f32
                 _Float16* __restrict__ h_state,                   // [2][B][512] f16
                 _Float16* __restrict__ x_next,                    // [B][T][1024] f16
                 int s)
{
  const int idx = blockIdx.x * 256 + threadIdx.x;  // over 2*B*512
  const int h = idx & 511;
  const int b = (idx >> 9) & (BB - 1);
  const int z = idx >> 19;
  const float* g  = gates + ((long long)z * BB + b) * 2048;
  const float* bi = z ? bih_b : bih_f;
  const float* bh = z ? bhh_b : bhh_f;
  const float ig = g[h]        + bi[h]        + bh[h];
  const float fg = g[512 + h]  + bi[512 + h]  + bh[512 + h];
  const float gg = g[1024 + h] + bi[1024 + h] + bh[1024 + h];
  const float og = g[1536 + h] + bi[1536 + h] + bh[1536 + h];
  float c = c_state[idx];
  c = sigm(fg) * c + sigm(ig) * tanhf(gg);
  const float hv = sigm(og) * tanhf(c);
  c_state[idx] = c;
  h_state[idx] = (_Float16)hv;
  const int te = z ? (TT - 1 - s) : s;
  x_next[((long long)b * TT + te) * 1024 + (long long)z * 512 + h] = (_Float16)hv;
}

__global__ void __launch_bounds__(256)
clear_state_kernel(_Float16* __restrict__ h, float* __restrict__ c, int n)
{
  const int i = blockIdx.x * 256 + threadIdx.x;
  if (i < n) { h[i] = (_Float16)0.f; c[i] = 0.f; }
}

__global__ void __launch_bounds__(256)
f32_to_f16_kernel(const float* __restrict__ in, _Float16* __restrict__ out, long long n)
{
  const long long i = (long long)blockIdx.x * 256 + threadIdx.x;
  if (i < n) out[i] = (_Float16)in[i];
}

// ---------------- Embeddings (Linear + LayerNorm, dim 64) ------------
__global__ void __launch_bounds__(64)
embed_seq_kernel(const float* __restrict__ seq,   // [B,T,18]
                 const float* __restrict__ W,     // [64,18]
                 const float* __restrict__ bias, const float* __restrict__ g,
                 const float* __restrict__ be, _Float16* __restrict__ x0)  // [B,T,128]
{
  const int row = blockIdx.x;         // b*T + t
  const int e = threadIdx.x;
  __shared__ float xin[18];
  __shared__ float y[64];
  if (e < 18) xin[e] = seq[(long long)row * 18 + e];
  __syncthreads();
  float acc = bias[e];
  const float* w = W + (long long)e * 18;
#pragma unroll
  for (int k = 0; k < 18; ++k) acc += xin[k] * w[k];
  y[e] = acc;
  __syncthreads();
  float mu = 0.f;
  for (int k = 0; k < 64; ++k) mu += y[k];
  mu *= (1.f / 64.f);
  float var = 0.f;
  for (int k = 0; k < 64; ++k) { float d = y[k] - mu; var += d * d; }
  var *= (1.f / 64.f);
  const float o = (acc - mu) * rsqrtf(var + 1e-5f) * g[e] + be[e];
  x0[(long long)row * 128 + e] = (_Float16)o;
}

__global__ void __launch_bounds__(64)
embed_perlevel_kernel(const float* __restrict__ other,  // [B,17]
                      const float* __restrict__ W,      // [60,64,17]
                      const float* __restrict__ bias,   // [60,64]
                      const float* __restrict__ g, const float* __restrict__ be, // [60,64]
                      _Float16* __restrict__ x0)        // [B,T,128] (cols 64..127)
{
  const int row = blockIdx.x;         // b*T + t
  const int t = row % TT;
  const int b = row / TT;
  const int e = threadIdx.x;
  __shared__ float xin[17];
  __shared__ float y[64];
  if (e < 17) xin[e] = other[(long long)b * 17 + e];
  __syncthreads();
  float acc = bias[(long long)t * 64 + e];
  const float* w = W + ((long long)t * 64 + e) * 17;
#pragma unroll
  for (int k = 0; k < 17; ++k) acc += xin[k] * w[k];
  y[e] = acc;
  __syncthreads();
  float mu = 0.f;
  for (int k = 0; k < 64; ++k) mu += y[k];
  mu *= (1.f / 64.f);
  float var = 0.f;
  for (int k = 0; k < 64; ++k) { float d = y[k] - mu; var += d * d; }
  var *= (1.f / 64.f);
  const float o = (acc - mu) * rsqrtf(var + 1e-5f) * g[(long long)t * 64 + e]
                + be[(long long)t * 64 + e];
  x0[(long long)row * 128 + 64 + e] = (_Float16)o;
}

__global__ void __launch_bounds__(64)
embed_other_kernel(const float* __restrict__ other,  // [B,17]
                   const float* __restrict__ W,      // [64,17]
                   const float* __restrict__ bias, const float* __restrict__ g,
                   const float* __restrict__ be, float* __restrict__ oe)  // [B,64] f32
{
  const int b = blockIdx.x;
  const int e = threadIdx.x;
  __shared__ float xin[17];
  __shared__ float y[64];
  if (e < 17) xin[e] = other[(long long)b * 17 + e];
  __syncthreads();
  float acc = bias[e];
  const float* w = W + (long long)e * 17;
#pragma unroll
  for (int k = 0; k < 17; ++k) acc += xin[k] * w[k];
  y[e] = acc;
  __syncthreads();
  float mu = 0.f;
  for (int k = 0; k < 64; ++k) mu += y[k];
  mu *= (1.f / 64.f);
  float var = 0.f;
  for (int k = 0; k < 64; ++k) { float d = y[k] - mu; var += d * d; }
  var *= (1.f / 64.f);
  oe[(long long)b * 64 + e] = (acc - mu) * rsqrtf(var + 1e-5f) * g[e] + be[e];
}

// ---------------- mean pool over T + LN(1024) + concat oe ------------
__global__ void __launch_bounds__(256)
pool_ln_kernel(const _Float16* __restrict__ x,   // [B,T,1024]
               const float* __restrict__ g, const float* __restrict__ be, // [1024]
               const float* __restrict__ oe,     // [B,64]
               _Float16* __restrict__ cat)       // [B,1088]
{
  const int b = blockIdx.x;
  const int tid = threadIdx.x;
  __shared__ float vals[1024];
  __shared__ float red[256];
  for (int j = tid; j < 1024; j += 256) {
    const _Float16* p = x + (long long)b * TT * 1024 + j;
    float s = 0.f;
    for (int t = 0; t < TT; ++t) s += (float)p[(long long)t * 1024];
    vals[j] = s * (1.f / (float)TT);
  }
  __syncthreads();
  float ls = 0.f, ls2 = 0.f;
  for (int j = tid; j < 1024; j += 256) { float v = vals[j]; ls += v; ls2 += v * v; }
  red[tid] = ls;
  __syncthreads();
  for (int off = 128; off > 0; off >>= 1) { if (tid < off) red[tid] += red[tid + off]; __syncthreads(); }
  const float mu = red[0] * (1.f / 1024.f);
  __syncthreads();
  red[tid] = ls2;
  __syncthreads();
  for (int off = 128; off > 0; off >>= 1) { if (tid < off) red[tid] += red[tid + off]; __syncthreads(); }
  const float var = red[0] * (1.f / 1024.f) - mu * mu;
  const float rs = rsqrtf(var + 1e-5f);
  for (int j = tid; j < 1024; j += 256)
    cat[(long long)b * 1088 + j] = (_Float16)((vals[j] - mu) * rs * g[j] + be[j]);
  if (tid < 64)
    cat[(long long)b * 1088 + 1024 + tid] = (_Float16)oe[(long long)b * 64 + tid];
}

// ------------- head epilogue: +b1, LN(256), ReLU, tiny matvec --------
__global__ void __launch_bounds__(256)
head_epi_kernel(const float* __restrict__ Y,    // [rows,256] (GEMM output)
                const float* __restrict__ b1, const float* __restrict__ g,
                const float* __restrict__ be,
                const float* __restrict__ W2,   // [nout,256]
                const float* __restrict__ b2,
                float* __restrict__ out, int nout)
{
  const int row = blockIdx.x;
  const int tid = threadIdx.x;
  __shared__ float hbuf[256];
  __shared__ float red[256];
  const float v = Y[(long long)row * 256 + tid] + b1[tid];
  red[tid] = v;
  __syncthreads();
  for (int off = 128; off > 0; off >>= 1) { if (tid < off) red[tid] += red[tid + off]; __syncthreads(); }
  const float mu = red[0] * (1.f / 256.f);
  __syncthreads();
  const float d = v - mu;
  red[tid] = d * d;
  __syncthreads();
  for (int off = 128; off > 0; off >>= 1) { if (tid < off) red[tid] += red[tid + off]; __syncthreads(); }
  const float var = red[0] * (1.f / 256.f);
  float hv = d * rsqrtf(var + 1e-5f) * g[tid] + be[tid];
  hv = fmaxf(hv, 0.f);
  hbuf[tid] = hv;
  __syncthreads();
  if (tid < nout) {
    float a = b2[tid];
    const float* w = W2 + (long long)tid * 256;
    for (int k = 0; k < 256; ++k) a += hbuf[k] * w[k];
    out[(long long)row * nout + tid] = a;
  }
}

// =====================================================================
// Host orchestration
// =====================================================================
extern "C" void kernel_launch(void* const* d_in, const int* in_sizes, int n_in,
                              void* d_out, int out_size, void* d_ws, size_t ws_size,
                              hipStream_t stream)
{
  (void)in_sizes; (void)n_in; (void)out_size; (void)ws_size;

  // d_in mapping: setup_inputs() insertion order, recursive.
  const float* seq     = (const float*)d_in[0];
  const float* other   = (const float*)d_in[1];
  const float* lin1_W  = (const float*)d_in[2];
  const float* lin1_b  = (const float*)d_in[3];
  const float* ln1_g   = (const float*)d_in[4];
  const float* ln1_b   = (const float*)d_in[5];
  const float* lin2s_W = (const float*)d_in[6];
  const float* lin2s_b = (const float*)d_in[7];
  const float* ln2s_g  = (const float*)d_in[8];
  const float* ln2s_b  = (const float*)d_in[9];
  const float* lin2_W  = (const float*)d_in[10];
  const float* lin2_b  = (const float*)d_in[11];
  const float* ln2_g   = (const float*)d_in[12];
  const float* ln2_b   = (const float*)d_in[13];
  const float* lnmid_g = (const float*)d_in[14];
  const float* lnmid_b = (const float*)d_in[15];
  const float* out1_W1 = (const float*)d_in[16];
  const float* out1_b1 = (const float*)d_in[17];
  const float* out1_g  = (const float*)d_in[18];
  const float* out1_bn = (const float*)d_in[19];
  const float* out1_W2 = (const float*)d_in[20];
  const float* out1_b2 = (const float*)d_in[21];
  const float* out2_W1 = (const float*)d_in[22];
  const float* out2_b1 = (const float*)d_in[23];
  const float* out2_g  = (const float*)d_in[24];
  const float* out2_bn = (const float*)d_in[25];
  const float* out2_W2 = (const float*)d_in[26];
  const float* out2_b2 = (const float*)d_in[27];
  // lstm layer l, base 28+8l: Wih_f, Whh_f, bih_f, bhh_f, Wih_b, Whh_b, bih_b, bhh_b

  const int B = BB, T = TT;
  const int Kin[3] = {128, 1024, 1024};

  // ---- carve workspace (256B aligned) ----
  char* wp = (char*)d_ws;
  auto carve = [&](size_t bytes) -> void* {
    void* r = (void*)wp;
    wp += (bytes + 255) & ~(size_t)255;
    return r;
  };
  _Float16* x0 = (_Float16*)carve((size_t)B * T * 128 * 2);
  _Float16* xA = (_Float16*)carve((size_t)B * T * 1024 * 2);
  _Float16* xB = (_Float16*)carve((size_t)B * T * 1024 * 2);
  _Float16* wih16[3];
  _Float16* whh16[3];
  for (int l = 0; l < 3; ++l) {
    wih16[l] = (_Float16*)carve((size_t)2 * 2048 * Kin[l] * 2);  // [dir][2048][Kin]
    whh16[l] = (_Float16*)carve((size_t)2 * 2048 * 512 * 2);     // [dir][2048][512]
  }
  _Float16* w1h16   = (_Float16*)carve((size_t)256 * 1024 * 2);
  _Float16* w2h16   = (_Float16*)carve((size_t)256 * 1088 * 2);
  _Float16* h_state = (_Float16*)carve((size_t)2 * B * 512 * 2);
  float*    c_state = (float*)carve((size_t)2 * B * 512 * 4);
  float*    gates   = (float*)carve((size_t)2 * B * 2048 * 4);
  float*    h1buf   = (float*)carve((size_t)B * T * 256 * 4);
  float*    h2buf   = (float*)carve((size_t)B * 256 * 4);
  float*    oe_buf  = (float*)carve((size_t)B * 64 * 4);
  _Float16* cat     = (_Float16*)carve((size_t)B * 1088 * 2);

  auto cvt = [&](const float* src, _Float16* dst, long long n) {
    int blocks = (int)((n + 255) / 256);
    f32_to_f16_kernel<<<blocks, 256, 0, stream>>>(src, dst, n);
  };

  // ---- convert weights to f16 ----
  for (int l = 0; l < 3; ++l) {
    const int base = 28 + l * 8;
    cvt((const float*)d_in[base + 0], wih16[l],                         (long long)2048 * Kin[l]);
    cvt((const float*)d_in[base + 4], wih16[l] + (size_t)2048 * Kin[l], (long long)2048 * Kin[l]);
    cvt((const float*)d_in[base + 1], whh16[l],                         (long long)2048 * 512);
    cvt((const float*)d_in[base + 5], whh16[l] + (size_t)2048 * 512,    (long long)2048 * 512);
  }
  cvt(out1_W1, w1h16, (long long)256 * 1024);
  cvt(out2_W1, w2h16, (long long)256 * 1088);

  // ---- embeddings -> x0 [B,T,128] f16, oe_buf [B,64] f32 ----
  embed_seq_kernel<<<B * T, 64, 0, stream>>>(seq, lin1_W, lin1_b, ln1_g, ln1_b, x0);
  embed_perlevel_kernel<<<B * T, 64, 0, stream>>>(other, lin2_W, lin2_b, ln2_g, ln2_b, x0);
  embed_other_kernel<<<B, 64, 0, stream>>>(other, lin2s_W, lin2s_b, ln2s_g, ln2s_b, oe_buf);

  // ---- 3-layer bidirectional LSTM ----
  for (int l = 0; l < 3; ++l) {
    const _Float16* xin = (l == 0) ? x0 : ((l == 1) ? xA : xB);
    _Float16* xout = (l == 0) ? xA : ((l == 1) ? xB : xA);
    const int K0 = Kin[l];
    const int base = 28 + l * 8;
    const float* bih_f = (const float*)d_in[base + 2];
    const float* bhh_f = (const float*)d_in[base + 3];
    const float* bih_b = (const float*)d_in[base + 6];
    const float* bhh_b = (const float*)d_in[base + 7];

    clear_state_kernel<<<(2 * B * 512) / 256, 256, 0, stream>>>(h_state, c_state, 2 * B * 512);

    for (int s = 0; s < T; ++s) {
      // z=0 (fwd) reads x[:,s,:]; z=1 (bwd) reads x[:,T-1-s,:] via signed z-offset
      const long long a0z = (long long)(T - 1 - 2 * s) * K0;
      dim3 gg(2048 / 128, B / 64, 2);
      wmma_gemm_kernel<<<gg, 128, 0, stream>>>(
          xin + (long long)s * K0, a0z, T * K0, K0,
          h_state, (long long)B * 512, 512, 512,
          wih16[l], (long long)2048 * K0,
          whh16[l], (long long)2048 * 512,
          gates, (long long)B * 2048, 2048);
      lstm_cell_kernel<<<(2 * B * 512) / 256, 256, 0, stream>>>(
          gates, bih_f, bhh_f, bih_b, bhh_b, c_state, h_state, xout, s);
    }
  }
  // final sequence output lives in xA: [B,T,1024] f16

  // ---- head 1: [B*T,1024] @ W1^T -> LN -> ReLU -> [.,6] ----
  {
    dim3 gg(256 / 128, (B * T) / 64, 1);
    wmma_gemm_kernel<<<gg, 128, 0, stream>>>(
        xA, 0, 1024, 1024,
        (const _Float16*)nullptr, 0, 0, 0,
        w1h16, 0, (const _Float16*)nullptr, 0,
        h1buf, 0, 256);
    head_epi_kernel<<<B * T, 256, 0, stream>>>(h1buf, out1_b1, out1_g, out1_bn,
                                               out1_W2, out1_b2, (float*)d_out, 6);
  }

  // ---- pooled path: mean over T + LN(1024) + concat oe -> [B,1088] ----
  pool_ln_kernel<<<B, 256, 0, stream>>>(xA, lnmid_g, lnmid_b, oe_buf, cat);

  // ---- head 2: [B,1088] @ W1^T -> LN -> ReLU -> [.,8] ----
  {
    dim3 gg(256 / 128, B / 64, 1);
    wmma_gemm_kernel<<<gg, 128, 0, stream>>>(
        cat, 0, 1088, 1088,
        (const _Float16*)nullptr, 0, 0, 0,
        w2h16, 0, (const _Float16*)nullptr, 0,
        h2buf, 0, 256);
    head_epi_kernel<<<B, 256, 0, stream>>>(h2buf, out2_b1, out2_g, out2_bn,
                                           out2_W2, out2_b2,
                                           (float*)d_out + (size_t)B * T * 6, 8);
  }
}
